// Encoder_6055903887925
// MI455X (gfx1250) — matrix operand order repaired and checked by reading, not compiled, not run on hardware
//
#include <hip/hip_runtime.h>
#include <hip/hip_bf16.h>
#include <math.h>

typedef __attribute__((ext_vector_type(16))) _Float16 v16h;
typedef __attribute__((ext_vector_type(8)))  float    v8f;

#define CH 512                   // points per compaction chunk (1 thread each)
#define WAVES_PER_BLOCK 8
#define SPLIT 8
#define WTOT (WAVES_PER_BLOCK * SPLIT)   // 64 waves per atom
#define OUT_RH 297               // 33*9
#define INV_CUTOFF 0.2f
#define PI_F 3.14159265358979323846f
#define SQRT2_F 1.41421356237309504880f

// Identical FP sequence everywhere so the selection predicate is consistent.
__device__ __forceinline__ unsigned dist_bits(float px, float py, float pz,
                                              const float* __restrict__ grid, int i,
                                              float* dx_o, float* dy_o, float* dz_o,
                                              float* ds2_o) {
  float gx = grid[3 * i + 0] * INV_CUTOFF;
  float gy = grid[3 * i + 1] * INV_CUTOFF;
  float gz = grid[3 * i + 2] * INV_CUTOFF;
  float dx = px - gx, dy = py - gy, dz = pz - gz;
  float ds2 = fmaf(dx, dx, fmaf(dy, dy, dz * dz));
  *dx_o = dx; *dy_o = dy; *dz_o = dz; *ds2_o = ds2;
  return __float_as_uint(ds2);
}

__device__ __forceinline__ unsigned dist_bits_only(float px, float py, float pz,
                                                   const float* __restrict__ grid, int i) {
  float a, b, c, d;
  return dist_bits(px, py, pz, grid, i, &a, &b, &c, &d);
}

__device__ __forceinline__ float shv(int n, float x, float y, float z) {
  const float s3 = 1.7320508075688772f;
  float r2 = x * x + y * y + z * z;
  switch (n) {
    case 0: return 1.0f;
    case 1: return y;
    case 2: return z;
    case 3: return x;
    case 4: return s3 * x * y;
    case 5: return s3 * y * z;
    case 6: return 0.5f * (3.0f * z * z - r2);
    case 7: return s3 * x * z;
    case 8: return 0.5f * s3 * (x * x - y * y);
    default: return 0.0f;
  }
}

// ---------------- K1: per-atom radix select of T-th smallest dist^2 bits -------
__global__ __launch_bounds__(256) void select_kernel(
    const float* __restrict__ nuc, const float* __restrict__ grid,
    int N, int T, unsigned* __restrict__ thrArr, unsigned* __restrict__ needArr) {
  int a = blockIdx.x;
  int tid = threadIdx.x;
  float px = nuc[3 * a + 0] * INV_CUTOFF;
  float py = nuc[3 * a + 1] * INV_CUTOFF;
  float pz = nuc[3 * a + 2] * INV_CUTOFF;

  __shared__ unsigned hist[256];
  __shared__ unsigned sPref;
  unsigned prefix = 0;
  unsigned k = (unsigned)(T - 1);   // rank of threshold (thread 0 only)
  unsigned cntLess = 0;             // thread 0 only

  for (int p = 3; p >= 0; --p) {
    for (int b = tid; b < 256; b += 256) hist[b] = 0u;
    __syncthreads();
    int shift = p * 8;
    for (int i = tid; i < N; i += 256) {
      unsigned bits = dist_bits_only(px, py, pz, grid, i);
      bool match = (p == 3) || ((bits >> (shift + 8)) == prefix);
      if (match) atomicAdd(&hist[(bits >> shift) & 255u], 1u);
    }
    __syncthreads();
    if (tid == 0) {
      unsigned cum = 0, b = 0;
      for (; b < 255u; ++b) {
        unsigned h = hist[b];
        if (cum + h > k) break;
        cum += h;
      }
      k -= cum;
      cntLess += cum;
      sPref = (prefix << 8) | b;
    }
    __syncthreads();
    prefix = sPref;
  }
  if (tid == 0) {
    thrArr[a] = prefix;
    needArr[a] = (unsigned)T - cntLess;  // #ties at threshold to accept
  }
}

// ---------------- K2a: per (atom,chunk) counts of less / equal -----------------
__global__ __launch_bounds__(256) void count_kernel(
    const float* __restrict__ nuc, const float* __restrict__ grid,
    const unsigned* __restrict__ thrArr, int N, int NC,
    unsigned* __restrict__ cntL, unsigned* __restrict__ cntE) {
  int chunk = blockIdx.x * blockDim.x + threadIdx.x;
  int a = blockIdx.y;
  if (chunk >= NC) return;
  unsigned thr = thrArr[a];
  float px = nuc[3 * a + 0] * INV_CUTOFF;
  float py = nuc[3 * a + 1] * INV_CUTOFF;
  float pz = nuc[3 * a + 2] * INV_CUTOFF;
  int base = chunk * CH;
  unsigned cl = 0, ce = 0;
  for (int j = 0; j < CH; ++j) {
    int i = base + j;
    if (i < N) {
      unsigned bits = dist_bits_only(px, py, pz, grid, i);
      cl += (bits < thr) ? 1u : 0u;
      ce += (bits == thr) ? 1u : 0u;
    }
  }
  cntL[(size_t)a * NC + chunk] = cl;
  cntE[(size_t)a * NC + chunk] = ce;
}

// ---------------- K2b: per-atom exclusive scan + list tail padding -------------
__global__ __launch_bounds__(256) void scan_pad_kernel(
    unsigned* __restrict__ cntL, unsigned* __restrict__ cntE,
    unsigned* __restrict__ lessTot, int* __restrict__ list,
    int A, int NC, int T, int Tpad) {
  int t = threadIdx.x;
  if (t >= A) return;
  unsigned runL = 0, runE = 0;
  for (int c = 0; c < NC; ++c) {
    unsigned l = cntL[(size_t)t * NC + c];
    unsigned e = cntE[(size_t)t * NC + c];
    cntL[(size_t)t * NC + c] = runL;
    cntE[(size_t)t * NC + c] = runE;
    runL += l;
    runE += e;
  }
  lessTot[t] = runL;
  for (int j = T; j < Tpad; ++j) list[(size_t)t * Tpad + j] = -1;
}

// ---------------- K2c: deterministic ordered compaction ------------------------
__global__ __launch_bounds__(256) void write_kernel(
    const float* __restrict__ nuc, const float* __restrict__ grid,
    const unsigned* __restrict__ thrArr, const unsigned* __restrict__ needArr,
    const unsigned* __restrict__ lessTot,
    const unsigned* __restrict__ cntL, const unsigned* __restrict__ cntE,
    int* __restrict__ list, int N, int NC, int Tpad) {
  int chunk = blockIdx.x * blockDim.x + threadIdx.x;
  int a = blockIdx.y;
  if (chunk >= NC) return;
  unsigned thr = thrArr[a];
  unsigned need = needArr[a];
  unsigned ltot = lessTot[a];
  unsigned posL = cntL[(size_t)a * NC + chunk];
  unsigned posE = cntE[(size_t)a * NC + chunk];
  float px = nuc[3 * a + 0] * INV_CUTOFF;
  float py = nuc[3 * a + 1] * INV_CUTOFF;
  float pz = nuc[3 * a + 2] * INV_CUTOFF;
  int* lst = list + (size_t)a * Tpad;
  int base = chunk * CH;
  for (int j = 0; j < CH; ++j) {
    int i = base + j;
    if (i < N) {
      unsigned bits = dist_bits_only(px, py, pz, grid, i);
      if (bits < thr) {
        lst[posL++] = i;
      } else if (bits == thr) {
        if (posE < need) lst[ltot + posE] = i;
        ++posE;
      }
    }
  }
}

// ---------------- K3: WMMA accumulation C[48x16] += A[48x32] * B[32x16] --------
__global__ __launch_bounds__(256) void wmma_kernel(
    const float* __restrict__ nuc, const float* __restrict__ grid,
    const float* __restrict__ wgt, const float* __restrict__ nvl,
    const int* __restrict__ list, float* __restrict__ part,
    int Tpad, int CT) {
  int a = blockIdx.y;
  int tid = threadIdx.x;
  int lane = tid & 31;
  int wid = tid >> 5;
  int gw = blockIdx.x * WAVES_PER_BLOCK + wid;  // 0..WTOT-1
  int m = lane & 15;        // A-matrix row within tile, also B column
  int gsel = lane >> 4;     // lane half-group
  float cc = (float)((m == 0) ? 16 : m) * PI_F;  // sincos frequency for this lane

  float px = nuc[3 * a + 0] * INV_CUTOFF;
  float py = nuc[3 * a + 1] * INV_CUTOFF;
  float pz = nuc[3 * a + 2] * INV_CUTOFF;

  __shared__ float sd[WAVES_PER_BLOCK][32];
  __shared__ float sq[WAVES_PER_BLOCK][32];
  __shared__ float sx[WAVES_PER_BLOCK][32];
  __shared__ float sy[WAVES_PER_BLOCK][32];
  __shared__ float sz[WAVES_PER_BLOCK][32];

  v8f c0 = {}, c1 = {}, c2 = {};
  const int* lst = list + (size_t)a * Tpad;

  for (int c = gw; c < CT; c += WTOT) {
    int idx = lst[c * 32 + lane];
    float d = 0.f, q = 0.f, x = 0.f, y = 0.f, z = 0.f;
    if (idx >= 0) {
      float dx, dy, dz, ds2;
      (void)dist_bits(px, py, pz, grid, idx, &dx, &dy, &dz, &ds2);
      d = __builtin_amdgcn_sqrtf(ds2);          // raw v_sqrt_f32, no fixup chain
      float env = 0.f;
      if (d < 1.f) {
        float d2 = d * d;
        float d6 = d2 * d2 * d2;
        env = 1.f - 28.f * d6 + 48.f * d6 * d - 21.f * d6 * d2;
      }
      q = SQRT2_F * env * nvl[idx] * wgt[idx];
      float inv = __builtin_amdgcn_rcpf(d + 1e-15f);  // raw v_rcp_f32
      x = dx * inv; y = dy * inv; z = dz * inv;
    }
    sd[wid][lane] = d; sq[wid][lane] = q;
    sx[wid][lane] = x; sy[wid][lane] = y; sz[wid][lane] = z;
    __builtin_amdgcn_wave_barrier();  // keep DS writes before DS reads (in-order within wave)

    v16h a0, a1, a2, bf;
#pragma unroll
    for (int v = 0; v < 8; ++v) {
#pragma unroll
      for (int h = 0; h < 2; ++h) {
        int e = 2 * v + h;
        {  // A fragment: K = ((v<4)?0:16) + 8*gsel + 2*(v&3) + h  (ISA 16-bit A 16x32 layout)
          int k = ((v < 4) ? 0 : 16) + 8 * gsel + 2 * (v & 3) + h;
          float dd = sd[wid][k], qq = sq[wid][k];
          float s, co;
          __sincosf(cc * dd, &s, &co);
          float b0 = (m == 0) ? 0.1f : s;       // row m      : const / sin(m*pi*d)
          float b1 = (m == 0) ? s : co;         // row m+16   : sin(16*pi*d) / cos(m*pi*d)
          float b2 = (m == 0) ? co : 0.f;       // row m+32   : cos(16*pi*d) / pad
          a0[e] = (_Float16)(b0 * qq);
          a1[e] = (_Float16)(b1 * qq);
          a2[e] = (_Float16)(b2 * qq);
        }
        {  // B fragment: K = ((e<8)?0:16) + 8*gsel + (e&7), column n = m (ISA 16-bit B 32x16 layout)
          int k = ((e < 8) ? 0 : 16) + 8 * gsel + (e & 7);
          bf[e] = (_Float16)shv(m, sx[wid][k], sy[wid][k], sz[wid][k]);
        }
      }
    }
    c0 = __builtin_amdgcn_wmma_f32_16x16x32_f16(false, a0, false, bf, (short)0, c0, false, false);
    c1 = __builtin_amdgcn_wmma_f32_16x16x32_f16(false, a1, false, bf, (short)0, c1, false, false);
    c2 = __builtin_amdgcn_wmma_f32_16x16x32_f16(false, a2, false, bf, (short)0, c2, false, false);
  }

  // Deterministic per-wave partial: part[a][gw][row][col]
  float* pp = part + ((size_t)a * WTOT + gw) * OUT_RH;
  int col = m;
#pragma unroll
  for (int r = 0; r < 8; ++r) {
    int row0 = 0 + r + 8 * gsel;
    int row1 = 16 + r + 8 * gsel;
    int row2 = 32 + r + 8 * gsel;
    if (col < 9) {
      pp[row0 * 9 + col] = c0[r];
      pp[row1 * 9 + col] = c1[r];
      if (row2 < 33) pp[row2 * 9 + col] = c2[r];
    }
  }
}

// ---------------- K4: deterministic reduction over waves -----------------------
__global__ __launch_bounds__(256) void reduce_kernel(
    const float* __restrict__ part, float* __restrict__ out, int total) {
  int i = blockIdx.x * blockDim.x + threadIdx.x;
  if (i >= total) return;
  int a = i / OUT_RH;
  int e = i - a * OUT_RH;
  float s = 0.f;
  const float* pp = part + (size_t)a * WTOT * OUT_RH + e;
  for (int w = 0; w < WTOT; ++w) s += pp[(size_t)w * OUT_RH];
  out[i] = s;
}

extern "C" void kernel_launch(void* const* d_in, const int* in_sizes, int n_in,
                              void* d_out, int out_size, void* d_ws, size_t ws_size,
                              hipStream_t stream) {
  (void)n_in; (void)out_size; (void)ws_size;
  const float* nuc  = (const float*)d_in[0];
  const float* grid = (const float*)d_in[2];
  const float* wgt  = (const float*)d_in[3];
  const float* nvl  = (const float*)d_in[4];

  int A = in_sizes[0] / 3;
  int N = in_sizes[3];
  long long Tll = (long long)8 * N / A;
  int T = (int)((Tll < N) ? Tll : N);
  int Tpad = (T + 31) & ~31;
  int CT = Tpad / 32;
  int NC = (N + CH - 1) / CH;

  char* w = (char*)d_ws;
  size_t off = 0;
  auto carve = [&](size_t bytes) { size_t o = off; off = (off + bytes + 255) & ~(size_t)255; return o; };
  unsigned* thrArr  = (unsigned*)(w + carve((size_t)A * 4));
  unsigned* needArr = (unsigned*)(w + carve((size_t)A * 4));
  unsigned* lessTot = (unsigned*)(w + carve((size_t)A * 4));
  unsigned* cntL    = (unsigned*)(w + carve((size_t)A * NC * 4));
  unsigned* cntE    = (unsigned*)(w + carve((size_t)A * NC * 4));
  int*      list    = (int*)(w + carve((size_t)A * Tpad * 4));
  float*    part    = (float*)(w + carve((size_t)A * WTOT * OUT_RH * 4));
  float*    out     = (float*)d_out;

  // 1) radix-select threshold per atom
  select_kernel<<<A, 256, 0, stream>>>(nuc, grid, N, T, thrArr, needArr);

  // 2) count / 3) scan+pad / 4) ordered compaction
  dim3 gcnt((NC + 255) / 256, A);
  count_kernel<<<gcnt, 256, 0, stream>>>(nuc, grid, thrArr, N, NC, cntL, cntE);
  scan_pad_kernel<<<1, 256, 0, stream>>>(cntL, cntE, lessTot, list, A, NC, T, Tpad);
  write_kernel<<<gcnt, 256, 0, stream>>>(nuc, grid, thrArr, needArr, lessTot,
                                         cntL, cntE, list, N, NC, Tpad);

  // 5) WMMA accumulation
  dim3 gw(SPLIT, A);
  wmma_kernel<<<gw, 256, 0, stream>>>(nuc, grid, wgt, nvl, list, part, Tpad, CT);

  // 6) deterministic reduction to output
  int total = A * OUT_RH;
  reduce_kernel<<<(total + 255) / 256, 256, 0, stream>>>(part, out, total);
}
